// MultiHeadAttention_24962349924517
// MI455X (gfx1250) — matrix operand order repaired and checked by reading, not compiled, not run on hardware
//
#include <hip/hip_runtime.h>
#include <cstdint>

// MHA for B=4, S=2048, d_model=1024, H=16, d_key=64 on gfx1250 (wave32, WMMA).
// bf16 matmuls with f32 accumulation via v_wmma_f32_16x16x32_bf16; f32 softmax.
// Attention K/V blocks staged to LDS with double-buffered
// global_load_async_to_lds_b128 (ASYNCcnt). Workspace usage: 72 MB.

#define B_DIM 4
#define S_DIM 2048
#define DM    1024
#define H_DIM 16
#define DK    64

typedef __attribute__((ext_vector_type(16))) __bf16 v16bf;
typedef __attribute__((ext_vector_type(8)))  __bf16 v8bf;
typedef __attribute__((ext_vector_type(8)))  float  v8f;
typedef __attribute__((ext_vector_type(4)))  float  v4f;

static __device__ __forceinline__ v16bf cat16(v8bf lo, v8bf hi) {
  v16bf r;
#pragma unroll
  for (int i = 0; i < 8; ++i) { r[i] = lo[i]; r[i + 8] = hi[i]; }
  return r;
}

// Async 16-byte global -> LDS copy (per lane). ldsPtr low 32 bits = LDS offset.
static __device__ __forceinline__ void async_cp16(void* ldsPtr, const void* gPtr) {
  unsigned dst = (unsigned)(uintptr_t)ldsPtr;
  unsigned long long src = (unsigned long long)(uintptr_t)gPtr;
  asm volatile("global_load_async_to_lds_b128 %0, %1, off"
               :: "v"(dst), "v"(src) : "memory");
}
static __device__ __forceinline__ void async_wait0() {
  asm volatile("s_wait_asynccnt 0x0" ::: "memory");
}

// ---------------------------------------------------------------- convert ---
__global__ void mha_cvt_bf16(const float* __restrict__ src,
                             __bf16* __restrict__ dst, int n) {
  int i = blockIdx.x * blockDim.x + threadIdx.x;
  if (i < n) dst[i] = (__bf16)src[i];
}

// ----------------------------------------------------------- projection -----
// C[r, c] = sum_d X[r,d] * W[c,d];  X f32 [8192,1024], W bf16 [1024,1024].
// mode 0: write bf16 to out[b][h][s][k]   (Q / K projection)
// mode 1: write bf16 to out[b][h][k][s]   (V projection, transposed)
__global__ __launch_bounds__(256) void mha_proj_gemm(
    const float* __restrict__ X, const __bf16* __restrict__ W,
    __bf16* __restrict__ out, int mode) {
  const int lane = threadIdx.x & 31;
  const int wave = threadIdx.x >> 5;
  const int hl   = lane >> 4;
  const int l16  = lane & 15;
  const int m0   = blockIdx.x * 128 + wave * 16;
  const int c0   = blockIdx.y * 64;

  v8f acc[4] = {};
  const float* xrow = X + (size_t)(m0 + l16) * DM;

  for (int d0 = 0; d0 < DM; d0 += 32) {
    const float* xp = xrow + d0 + 8 * hl;
    v4f x0 = *(const v4f*)(xp);
    v4f x1 = *(const v4f*)(xp + 4);
    v4f x2 = *(const v4f*)(xp + 16);
    v4f x3 = *(const v4f*)(xp + 20);
    v16bf a;
#pragma unroll
    for (int i = 0; i < 4; ++i) {
      a[i]      = (__bf16)x0[i];
      a[4 + i]  = (__bf16)x1[i];
      a[8 + i]  = (__bf16)x2[i];
      a[12 + i] = (__bf16)x3[i];
    }
#pragma unroll
    for (int t = 0; t < 4; ++t) {
      const __bf16* wr = W + (size_t)(c0 + 16 * t + l16) * DM + d0 + 8 * hl;
      v16bf b = cat16(*(const v8bf*)wr, *(const v8bf*)(wr + 16));
      acc[t] = __builtin_amdgcn_wmma_f32_16x16x32_bf16(
          false, a, false, b, (short)0, acc[t], false, false);
    }
  }

#pragma unroll
  for (int t = 0; t < 4; ++t) {
    const int col = c0 + 16 * t + l16;
    const int h = col >> 6, k = col & 63;
#pragma unroll
    for (int v = 0; v < 8; ++v) {
      const int r = m0 + v + 8 * hl;
      const int b = r >> 11, s = r & (S_DIM - 1);
      size_t off;
      if (mode == 0)
        off = (((size_t)(b * H_DIM + h)) * S_DIM + s) * DK + k;
      else
        off = (((size_t)(b * H_DIM + h)) * DK + k) * S_DIM + s;
      out[off] = (__bf16)acc[t][v];
    }
  }
}

// ------------------------------------------------------------- attention ----
// One workgroup: 128 query rows of one (b,h). Wave owns 16 rows.
// K/V key-blocks (32 keys) are async-copied into double-buffered LDS tiles,
// shared by all 8 waves; online softmax in f32; O accumulated in f32.
__global__ __launch_bounds__(256) void mha_attn(
    const __bf16* __restrict__ Qp,   // [B,H,S,64]
    const __bf16* __restrict__ Kp,   // [B,H,S,64]
    const __bf16* __restrict__ VpT,  // [B,H,64,S]
    __bf16* __restrict__ AO) {       // [B,S,1024]
  __shared__ __align__(16) __bf16 klds[2][32][64];   // [buf][key][dk]
  __shared__ __align__(16) __bf16 vlds[2][64][32];   // [buf][dk][key]
  __shared__ __align__(16) __bf16 pbuf[8][16][32];   // per-wave P reshape

  const int tid  = threadIdx.x;
  const int lane = tid & 31, wave = tid >> 5;
  const int hl = lane >> 4, l16 = lane & 15;
  const int bh = blockIdx.y;               // b*16 + h
  const int q0 = blockIdx.x * 128 + wave * 16;

  const __bf16* kbase = Kp + (size_t)bh * S_DIM * DK;
  const __bf16* vbase = VpT + (size_t)bh * DK * S_DIM;

  // cooperative async-copy mapping: 256 threads x 16B cover each 4KB tile
  const int krow = tid >> 3, kch = tid & 7;   // K tile: 32 rows x 8 chunks
  const int vrow = tid >> 2, vch = tid & 3;   // V tile: 64 rows x 4 chunks

  auto issue_block = [&](int buf, int kb) {
    async_cp16(&klds[buf][krow][kch * 8],
               kbase + (size_t)(kb + krow) * DK + kch * 8);
    async_cp16(&vlds[buf][vrow][vch * 8],
               vbase + (size_t)vrow * S_DIM + kb + vch * 8);
  };

  issue_block(0, 0);  // prefetch first key block

  // Q A-fragments for both 32-wide contraction chunks over d_key
  const __bf16* qrow = Qp + ((size_t)bh * S_DIM + q0 + l16) * DK;
  v16bf qa[2];
#pragma unroll
  for (int c = 0; c < 2; ++c) {
    const __bf16* qp_ = qrow + 32 * c + 16 * hl;
    qa[c] = cat16(*(const v8bf*)qp_, *(const v8bf*)(qp_ + 8));
  }

  v8f o[4] = {};
  float rm[8], rs[8];
#pragma unroll
  for (int v = 0; v < 8; ++v) { rm[v] = -__builtin_inff(); rs[v] = 0.f; }

  __bf16* pb = &pbuf[wave][0][0];

  async_wait0();
  __syncthreads();

  const int NB = S_DIM / 32;
  for (int ib = 0; ib < NB; ++ib) {
    const int cur = ib & 1;
    if (ib + 1 < NB) issue_block(cur ^ 1, (ib + 1) * 32);

    // logits tile 16 x 32 from LDS K tile
    v8f c0 = {}, c1 = {};
#pragma unroll
    for (int c = 0; c < 2; ++c) {
      const __bf16* kr0 = &klds[cur][l16][32 * c + 16 * hl];
      v16bf b0 = cat16(*(const v8bf*)kr0, *(const v8bf*)(kr0 + 8));
      c0 = __builtin_amdgcn_wmma_f32_16x16x32_bf16(
          false, qa[c], false, b0, (short)0, c0, false, false);
      const __bf16* kr1 = &klds[cur][16 + l16][32 * c + 16 * hl];
      v16bf b1 = cat16(*(const v8bf*)kr1, *(const v8bf*)(kr1 + 8));
      c1 = __builtin_amdgcn_wmma_f32_16x16x32_bf16(
          false, qa[c], false, b1, (short)0, c1, false, false);
    }

    // online softmax per row (row = v + 8*hl, spread over 16 lanes of a half)
    float al[8];
#pragma unroll
    for (int v = 0; v < 8; ++v) {
      float x0 = c0[v] * 0.125f;            // 1/sqrt(64)
      float x1 = c1[v] * 0.125f;
      float t = fmaxf(x0, x1);
#pragma unroll
      for (int m2 = 1; m2 < 16; m2 <<= 1) t = fmaxf(t, __shfl_xor(t, m2, 32));
      float nm = fmaxf(rm[v], t);
      al[v] = __expf(rm[v] - nm);
      rm[v] = nm;
      float p0 = __expf(x0 - nm), p1 = __expf(x1 - nm);
      float sum = p0 + p1;
#pragma unroll
      for (int m2 = 1; m2 < 16; m2 <<= 1) sum += __shfl_xor(sum, m2, 32);
      rs[v] = rs[v] * al[v] + sum;
      pb[(v + 8 * hl) * 32 + l16]      = (__bf16)p0;
      pb[(v + 8 * hl) * 32 + 16 + l16] = (__bf16)p1;
    }
    __syncthreads();
    // reload P as a WMMA A-fragment (row = l16, K = 16*hl..16*hl+15)
    const __bf16* pr = pb + l16 * 32 + 16 * hl;
    v16bf pa = cat16(*(const v8bf*)pr, *(const v8bf*)(pr + 8));

#pragma unroll
    for (int t = 0; t < 4; ++t) {
      v8f ot = o[t];
#pragma unroll
      for (int v = 0; v < 8; ++v) ot[v] *= al[v];
      const __bf16* vr = &vlds[cur][16 * t + l16][16 * hl];
      v16bf vb = cat16(*(const v8bf*)vr, *(const v8bf*)(vr + 8));
      o[t] = __builtin_amdgcn_wmma_f32_16x16x32_bf16(
          false, pa, false, vb, (short)0, ot, false, false);
    }

    async_wait0();     // next block's tiles landed (and our own issues done)
    __syncthreads();   // everyone done reading buf[cur] and pbuf
  }

  const int b = bh >> 4, h = bh & 15;
#pragma unroll
  for (int t = 0; t < 4; ++t) {
#pragma unroll
    for (int v = 0; v < 8; ++v) {
      const int s = q0 + v + 8 * hl;
      size_t off = ((size_t)(b * S_DIM + s)) * DM + h * DK + 16 * t + l16;
      AO[off] = (__bf16)(o[t][v] / rs[v]);
    }
  }
}

// ----------------------------------------------------------- output GEMM ----
// out[r,c] = sum_d AO[r,d] * Wo[c,d];  AO bf16 [8192,1024] -> f32 out.
__global__ __launch_bounds__(256) void mha_out_gemm(
    const __bf16* __restrict__ A, const __bf16* __restrict__ W,
    float* __restrict__ out) {
  const int lane = threadIdx.x & 31;
  const int wave = threadIdx.x >> 5;
  const int hl = lane >> 4, l16 = lane & 15;
  const int m0 = blockIdx.x * 128 + wave * 16;
  const int c0 = blockIdx.y * 64;

  v8f acc[4] = {};
  const __bf16* arow = A + (size_t)(m0 + l16) * DM;

  for (int d0 = 0; d0 < DM; d0 += 32) {
    const __bf16* ap = arow + d0 + 16 * hl;
    v16bf a = cat16(*(const v8bf*)ap, *(const v8bf*)(ap + 8));
#pragma unroll
    for (int t = 0; t < 4; ++t) {
      const __bf16* wr = W + (size_t)(c0 + 16 * t + l16) * DM + d0 + 16 * hl;
      v16bf b = cat16(*(const v8bf*)wr, *(const v8bf*)(wr + 8));
      acc[t] = __builtin_amdgcn_wmma_f32_16x16x32_bf16(
          false, a, false, b, (short)0, acc[t], false, false);
    }
  }
#pragma unroll
  for (int t = 0; t < 4; ++t) {
    const int col = c0 + 16 * t + l16;
#pragma unroll
    for (int v = 0; v < 8; ++v)
      out[(size_t)(m0 + v + 8 * hl) * DM + col] = acc[t][v];
  }
}

// ------------------------------------------------------------------ launch --
extern "C" void kernel_launch(void* const* d_in, const int* in_sizes, int n_in,
                              void* d_out, int out_size, void* d_ws,
                              size_t ws_size, hipStream_t stream) {
  const float* queries = (const float*)d_in[0];
  const float* keys    = (const float*)d_in[1];
  const float* values  = (const float*)d_in[2];
  const float* Wq      = (const float*)d_in[3];
  const float* Wk      = (const float*)d_in[4];
  const float* Wv      = (const float*)d_in[5];
  const float* Wo      = (const float*)d_in[6];
  float* out = (float*)d_out;

  // workspace layout (bf16 elements); total 36M elems = 72 MB
  __bf16* wq  = (__bf16*)d_ws;
  __bf16* wk  = wq  + (1u << 20);
  __bf16* wv  = wk  + (1u << 20);
  __bf16* wo  = wv  + (1u << 20);
  __bf16* Qp  = wo  + (1u << 20);   // [B,H,S,64]
  __bf16* Kp  = Qp  + (8u << 20);   // [B,H,S,64]
  __bf16* VpT = Kp  + (8u << 20);   // [B,H,64,S]
  __bf16* AO  = VpT + (8u << 20);   // [B,S,1024]

  const int NW = 1 << 20;           // 1024*1024 weight elems per matrix
  mha_cvt_bf16<<<NW / 256, 256, 0, stream>>>(Wq, wq, NW);
  mha_cvt_bf16<<<NW / 256, 256, 0, stream>>>(Wk, wk, NW);
  mha_cvt_bf16<<<NW / 256, 256, 0, stream>>>(Wv, wv, NW);
  mha_cvt_bf16<<<NW / 256, 256, 0, stream>>>(Wo, wo, NW);

  dim3 g(64, 16), blk(256);
  mha_proj_gemm<<<g, blk, 0, stream>>>(queries, wq, Qp, 0);
  mha_proj_gemm<<<g, blk, 0, stream>>>(keys,    wk, Kp, 0);
  mha_proj_gemm<<<g, blk, 0, stream>>>(values,  wv, VpT, 1);

  mha_attn<<<dim3(16, 64), blk, 0, stream>>>(Qp, Kp, VpT, AO);

  mha_out_gemm<<<g, blk, 0, stream>>>(AO, wo, out);
}
